// TransformerBlockInnerLayer_46986942218961
// MI455X (gfx1250) — matrix operand order repaired and checked by reading, not compiled, not run on hardware
//
#include <hip/hip_runtime.h>
#include <hip/hip_bf16.h>
#include <cstdint>
#include <cstddef>

// ---------------------------------------------------------------------------
// Types for CDNA5 WMMA (wave32): bf16 A/B fragments (16 VGPR-packed elems),
// f32 8-VGPR accumulators.
// ---------------------------------------------------------------------------
typedef __bf16 bf16;
typedef bf16  bf16x8  __attribute__((ext_vector_type(8)));
typedef bf16  bf16x16 __attribute__((ext_vector_type(16)));
typedef float f32x4   __attribute__((ext_vector_type(4)));
typedef float f32x8   __attribute__((ext_vector_type(8)));

__device__ __forceinline__ f32x8 wmma_bf16(bf16x16 a, bf16x16 b, f32x8 c) {
  // D = A(16x32) * B(32x16) + C(16x16), f32 accumulate
  return __builtin_amdgcn_wmma_f32_16x16x32_bf16(
      /*neg_a=*/false, a, /*neg_b=*/false, b,
      /*c_mod=*/(short)0, c, /*reuse_a=*/false, /*reuse_b=*/false);
}

#define D_MODEL 1024
#define NHEADS  16
#define HD      64
#define LX      2048
#define LC      4096
#define HIDDEN  2816
#define NTOK    (LX + LC)          // 6144 rows through LN1/QKV
#define NROWS2  4096               // x-path (2048) + ctx-slice path (2048)

// ===========================================================================
// LayerNorm: one block (256 threads) per row of 1024 elements.
// rows < split read src1[row], else src2[row - split_off] (we pass split_off
// implicitly: ctx rows use src2 indexed by (row - split)).
// ===========================================================================
__global__ __launch_bounds__(256)
void ln_kernel(const float* __restrict__ src1, const float* __restrict__ src2,
               int split, const float* __restrict__ g, const float* __restrict__ b,
               float* __restrict__ out) {
  const int row = blockIdx.x;
  const int t = threadIdx.x;
  const float* src = (row < split) ? (src1 + (size_t)row * D_MODEL)
                                   : (src2 + (size_t)(row - split) * D_MODEL);
  f32x4 v = *(const f32x4*)(src + t * 4);

  __shared__ float red[256];
  float s = v[0] + v[1] + v[2] + v[3];
  red[t] = s; __syncthreads();
  for (int o = 128; o > 0; o >>= 1) { if (t < o) red[t] += red[t + o]; __syncthreads(); }
  const float mean = red[0] * (1.0f / D_MODEL);
  __syncthreads();

  float d0 = v[0] - mean, d1 = v[1] - mean, d2 = v[2] - mean, d3 = v[3] - mean;
  red[t] = d0 * d0 + d1 * d1 + d2 * d2 + d3 * d3; __syncthreads();
  for (int o = 128; o > 0; o >>= 1) { if (t < o) red[t] += red[t + o]; __syncthreads(); }
  const float rinv = rsqrtf(red[0] * (1.0f / D_MODEL) + 1e-5f);

  float* dst = out + (size_t)row * D_MODEL + t * 4;
  const float* gp = g + t * 4;
  const float* bp = b + t * 4;
  dst[0] = d0 * rinv * gp[0] + bp[0];
  dst[1] = d1 * rinv * gp[1] + bp[1];
  dst[2] = d2 * rinv * gp[2] + bp[2];
  dst[3] = d3 * rinv * gp[3] + bp[3];
}

// ===========================================================================
// Generic WMMA GEMM: C[M,N] = A[M,K](f32,row-major) * B[K,N](f32,row-major)
//                    (+ bias[N]) (+ Res[M,N]).
// fp32 is converted to bf16 while staging tiles into LDS (TF32-style).
// Block: 256 threads = 8 waves; macro tile 128x128, BK=32.
// Wave (wm in 0..3, wn in 0..1) owns a 32x64 sub-tile = 2x4 WMMA tiles.
// LDS holds A (128x32) and B^T (128x32) bf16, row stride 40 (16B-aligned pad).
// ===========================================================================
#define BM 128
#define BN 128
#define BK 32
#define LDT 40   // LDS tile row stride in elements (80 bytes, 16B aligned)

__global__ __launch_bounds__(256)
void gemm_bf16_kernel(const float* __restrict__ A, const float* __restrict__ B,
                      float* __restrict__ C, const float* __restrict__ bias,
                      const float* __restrict__ Res, int M, int N, int K) {
  __shared__ __align__(32) bf16 As[BM * LDT];
  __shared__ __align__(32) bf16 Bs[BN * LDT];

  const int tid  = threadIdx.x;
  const int lane = tid & 31;
  const int wid  = tid >> 5;
  const int wm   = wid & 3;        // 4 row slabs of 32
  const int wn   = wid >> 2;       // 2 col slabs of 64
  const int half = lane >> 4;      // 0/1 (wave32 halves)
  const int l16  = lane & 15;
  const int bm0  = blockIdx.x * BM;
  const int bn0  = blockIdx.y * BN;

  f32x8 acc[2][4];
#pragma unroll
  for (int mi = 0; mi < 2; ++mi)
#pragma unroll
    for (int ni = 0; ni < 4; ++ni)
      acc[mi][ni] = (f32x8){0.f, 0.f, 0.f, 0.f, 0.f, 0.f, 0.f, 0.f};

  const int ar = tid >> 3;          // 0..31 (x4 passes -> 128 rows)
  const int ac = (tid & 7) * 4;     // 0..28
  const int bk = tid >> 5;          // 0..7  (x4 passes -> 32 k rows)
  const int bn = (tid & 31) * 4;    // 0..124

  for (int k0 = 0; k0 < K; k0 += BK) {
    // ---- stage A 128x32 (f32 -> bf16) ----
#pragma unroll
    for (int p = 0; p < 4; ++p) {
      const int row = ar + p * 32;
      f32x4 v = *(const f32x4*)(A + (size_t)(bm0 + row) * K + k0 + ac);
      bf16* d = &As[row * LDT + ac];
      d[0] = (bf16)v[0]; d[1] = (bf16)v[1]; d[2] = (bf16)v[2]; d[3] = (bf16)v[3];
    }
    // ---- stage B^T 128x32 (f32 -> bf16, transposed) ----
#pragma unroll
    for (int p = 0; p < 4; ++p) {
      const int kr = bk + p * 8;
      f32x4 v = *(const f32x4*)(B + (size_t)(k0 + kr) * N + bn0 + bn);
      Bs[(bn + 0) * LDT + kr] = (bf16)v[0];
      Bs[(bn + 1) * LDT + kr] = (bf16)v[1];
      Bs[(bn + 2) * LDT + kr] = (bf16)v[2];
      Bs[(bn + 3) * LDT + kr] = (bf16)v[3];
    }
    if (k0 + BK < K) {  // gfx1250 global_prefetch_b8 of next K-slab
      __builtin_prefetch(A + (size_t)(bm0 + ar) * K + k0 + BK + ac, 0, 1);
      __builtin_prefetch(B + (size_t)(k0 + BK + bk) * N + bn0 + bn, 0, 1);
    }
    __syncthreads();

    // ---- fragments + 8 WMMAs ----
    union { bf16x16 f; bf16x8 h8[2]; } afr[2];
    bf16x16 bfr[4];
#pragma unroll
    for (int mi = 0; mi < 2; ++mi) {
      const int row = wm * 32 + mi * 16 + l16;
      const int kb  = half * 16;     // contiguous k = 16*half + s (matches B)
      afr[mi].h8[0] = *(const bf16x8*)&As[row * LDT + kb];
      afr[mi].h8[1] = *(const bf16x8*)&As[row * LDT + kb + 8];
    }
#pragma unroll
    for (int ni = 0; ni < 4; ++ni) {
      const int col = wn * 64 + ni * 16 + l16;
      bfr[ni] = *(const bf16x16*)&Bs[col * LDT + half * 16];  // B 32x16 layout
    }
#pragma unroll
    for (int mi = 0; mi < 2; ++mi)
#pragma unroll
      for (int ni = 0; ni < 4; ++ni)
        acc[mi][ni] = wmma_bf16(afr[mi].f, bfr[ni], acc[mi][ni]);
    __syncthreads();
  }

  // ---- epilogue: C(16x16) layout -> row = r + half*8, col = l16 ----
#pragma unroll
  for (int mi = 0; mi < 2; ++mi)
#pragma unroll
    for (int ni = 0; ni < 4; ++ni) {
      const int col = bn0 + wn * 64 + ni * 16 + l16;
      const float bv = bias ? bias[col] : 0.f;
#pragma unroll
      for (int r = 0; r < 8; ++r) {
        const int row = bm0 + wm * 32 + mi * 16 + half * 8 + r;
        float v = acc[mi][ni][r] + bv;
        if (Res) v += Res[(size_t)row * N + col];
        C[(size_t)row * N + col] = v;
      }
    }
}

// ===========================================================================
// RoPE + head split + bf16 convert. V stored transposed per head: Vt[h][d][L]
// so PV WMMA B-fragments are contiguous 32B loads.
// Block 256 = 8 rows x 32 lanes; lane j handles rotation pair (j, j+32).
// ===========================================================================
__global__ __launch_bounds__(256)
void rope_kernel(const float* __restrict__ qkv,
                 bf16* __restrict__ Qx, bf16* __restrict__ Kx, bf16* __restrict__ Vxt,
                 bf16* __restrict__ Qc, bf16* __restrict__ Kc, bf16* __restrict__ Vct,
                 const int* __restrict__ xpos_p, const int* __restrict__ cpos_p) {
  const int j   = threadIdx.x & 31;
  const int sub = threadIdx.x >> 5;
  const int row = blockIdx.x * 8 + sub;      // 0..6143
  const int h   = blockIdx.y;

  const float* p = qkv + (size_t)row * (3 * D_MODEL) + h * HD;
  const float q1 = p[j],              q2 = p[j + 32];
  const float k1 = p[D_MODEL + j],    k2 = p[D_MODEL + j + 32];
  const float v1 = p[2*D_MODEL + j],  v2 = p[2*D_MODEL + j + 32];

  const bool isx = (row < LX);
  const int lrow = isx ? row : row - LX;
  const int pos  = isx ? (xpos_p[0] + lrow) : (cpos_p[0] + lrow);
  const int L    = isx ? LX : LC;

  const float inv = __powf(10000.0f, -(float)j * (1.0f / 32.0f));
  float s, c;
  __sincosf((float)pos * inv, &s, &c);

  bf16* Qd = (isx ? Qx : Qc) + ((size_t)h * L + lrow) * HD;
  bf16* Kd = (isx ? Kx : Kc) + ((size_t)h * L + lrow) * HD;
  bf16* Vd = (isx ? Vxt : Vct) + (size_t)h * HD * L;

  Qd[j]      = (bf16)(q1 * c - q2 * s);
  Qd[j + 32] = (bf16)(q2 * c + q1 * s);
  Kd[j]      = (bf16)(k1 * c - k2 * s);
  Kd[j + 32] = (bf16)(k2 * c + k1 * s);
  Vd[(size_t)j * L + lrow]        = (bf16)v1;
  Vd[(size_t)(j + 32) * L + lrow] = (bf16)v2;
}

// ===========================================================================
// Flash attention, one wave per (head, 16-row q tile); 4 waves / block.
// QK^T: 2 chained v_wmma_f32_16x16x32_bf16 over HD=64, K fragments loaded
// straight from global (rows contiguous). Online softmax via wave32 half-wave
// __shfl_xor reductions. P re-laid-out C->A through a private LDS tile.
// PV: V pre-transposed, B fragments are contiguous 32B loads.
// ===========================================================================
#define PLT 40  // P LDS row stride
__global__ __launch_bounds__(128)
void attn_kernel(const bf16* __restrict__ Q, const bf16* __restrict__ K,
                 const bf16* __restrict__ Vt, float* __restrict__ Out,
                 int Lk, long qsH, long ksH, int causal, int accum) {
  __shared__ __align__(32) bf16 Pl[4][16 * PLT];

  const int lane  = threadIdx.x & 31;
  const int w     = threadIdx.x >> 5;
  const int half  = lane >> 4;
  const int l16   = lane & 15;
  const int h     = blockIdx.y;
  const int qtile = blockIdx.x * 4 + w;

  const bf16* Qb = Q + (size_t)h * qsH + (size_t)qtile * 16 * HD;
  const bf16* Kb = K + (size_t)h * ksH;
  const bf16* Vb = Vt + (size_t)h * ksH;
  bf16* Pw = &Pl[w][0];

  union { bf16x16 f; bf16x8 h8[2]; } qf[2];
#pragma unroll
  for (int ki = 0; ki < 2; ++ki) {
    const bf16* src = Qb + l16 * HD + ki * 32 + half * 16;
    qf[ki].h8[0] = *(const bf16x8*)src;
    qf[ki].h8[1] = *(const bf16x8*)(src + 8);
  }

  f32x8 o[4];
#pragma unroll
  for (int ni = 0; ni < 4; ++ni)
    o[ni] = (f32x8){0.f, 0.f, 0.f, 0.f, 0.f, 0.f, 0.f, 0.f};
  float mrun[8], lrun[8];
#pragma unroll
  for (int v = 0; v < 8; ++v) { mrun[v] = -3.0e38f; lrun[v] = 0.f; }

  for (int kb0 = 0; kb0 < Lk; kb0 += 32) {
    f32x8 s0 = (f32x8){0.f,0.f,0.f,0.f,0.f,0.f,0.f,0.f};
    f32x8 s1 = (f32x8){0.f,0.f,0.f,0.f,0.f,0.f,0.f,0.f};
#pragma unroll
    for (int ki = 0; ki < 2; ++ki) {
      bf16x16 kf0 = *(const bf16x16*)(Kb + (size_t)(kb0 + l16) * HD + ki * 32 + half * 16);
      bf16x16 kf1 = *(const bf16x16*)(Kb + (size_t)(kb0 + 16 + l16) * HD + ki * 32 + half * 16);
      s0 = wmma_bf16(qf[ki].f, kf0, s0);
      s1 = wmma_bf16(qf[ki].f, kf1, s1);
    }
#pragma unroll
    for (int v = 0; v < 8; ++v) {
      float a = s0[v] * 0.125f;     // 1/sqrt(64)
      float b = s1[v] * 0.125f;
      if (causal) {
        const int qi = qtile * 16 + v + half * 8;
        if (kb0 + l16 > qi)      a = -3.0e38f;
        if (kb0 + 16 + l16 > qi) b = -3.0e38f;
      }
      float t = fmaxf(a, b);
      t = fmaxf(t, __shfl_xor(t, 1));
      t = fmaxf(t, __shfl_xor(t, 2));
      t = fmaxf(t, __shfl_xor(t, 4));
      t = fmaxf(t, __shfl_xor(t, 8));        // row max within 16-lane half
      const float mn = fmaxf(mrun[v], t);
      const float alpha = __expf(mrun[v] - mn);
      mrun[v] = mn;
      const float p0 = __expf(a - mn);
      const float p1 = __expf(b - mn);
      float rs = p0 + p1;
      rs += __shfl_xor(rs, 1);
      rs += __shfl_xor(rs, 2);
      rs += __shfl_xor(rs, 4);
      rs += __shfl_xor(rs, 8);
      lrun[v] = lrun[v] * alpha + rs;
      o[0][v] *= alpha; o[1][v] *= alpha; o[2][v] *= alpha; o[3][v] *= alpha;
      Pw[(v + half * 8) * PLT + l16]      = (bf16)p0;  // C-layout -> LDS
      Pw[(v + half * 8) * PLT + 16 + l16] = (bf16)p1;
    }
    // re-read P as an A fragment (DS ops from same wave are in-order)
    union { bf16x16 f; bf16x8 h8[2]; } pf;
    pf.h8[0] = *(const bf16x8*)(Pw + l16 * PLT + half * 16);
    pf.h8[1] = *(const bf16x8*)(Pw + l16 * PLT + half * 16 + 8);
#pragma unroll
    for (int ni = 0; ni < 4; ++ni) {
      bf16x16 vf = *(const bf16x16*)(Vb + (size_t)(ni * 16 + l16) * Lk + kb0 + half * 16);
      o[ni] = wmma_bf16(pf.f, vf, o[ni]);
    }
  }

#pragma unroll
  for (int ni = 0; ni < 4; ++ni)
#pragma unroll
    for (int v = 0; v < 8; ++v) {
      const int row = qtile * 16 + v + half * 8;
      const int col = h * HD + ni * 16 + l16;
      float* dst = Out + (size_t)row * D_MODEL + col;
      const float val = o[ni][v] / lrun[v];
      if (accum) *dst += val; else *dst = val;
    }
}

// ===========================================================================
// Elementwise kernels
// ===========================================================================
__global__ __launch_bounds__(256)
void silu_mul_kernel(const float* __restrict__ h1, const float* __restrict__ h3,
                     float* __restrict__ hg) {
  const size_t i = (size_t)blockIdx.x * 256 + threadIdx.x;
  const float x = h1[i];
  hg[i] = (x / (1.f + __expf(-x))) * h3[i];
}

// xo = proj + residual (rows<2048: x ; rows>=2048: context rows 2048..4095)
__global__ __launch_bounds__(256)
void add_res_kernel(const float* __restrict__ proj, const float* __restrict__ x,
                    const float* __restrict__ ctx, float* __restrict__ xo) {
  const size_t i = (size_t)blockIdx.x * 256 + threadIdx.x;
  const size_t row = i >> 10;
  const float r = (row < LX) ? x[i] : ctx[i];
  xo[i] = proj[i] + r;
}

// ===========================================================================
// Host orchestration
// ===========================================================================
static inline void launch_gemm(const float* A, const float* B, float* C,
                               const float* bias, const float* res,
                               int M, int N, int K, hipStream_t s) {
  dim3 g(M / BM, N / BN), b(256);
  gemm_bf16_kernel<<<g, b, 0, s>>>(A, B, C, bias, res, M, N, K);
}

extern "C" void kernel_launch(void* const* d_in, const int* in_sizes, int n_in,
                              void* d_out, int out_size, void* d_ws, size_t ws_size,
                              hipStream_t stream) {
  const float* x     = (const float*)d_in[0];
  const float* ctx   = (const float*)d_in[1];
  const float* qkv_w = (const float*)d_in[2];
  const float* qkv_b = (const float*)d_in[3];
  const float* out_w = (const float*)d_in[4];
  const float* out_b = (const float*)d_in[5];
  const float* w1    = (const float*)d_in[6];
  const float* w3    = (const float*)d_in[7];
  const float* w2    = (const float*)d_in[8];
  const float* n1_g  = (const float*)d_in[9];
  const float* n1_b  = (const float*)d_in[10];
  const float* n2_g  = (const float*)d_in[11];
  const float* n2_b  = (const float*)d_in[12];
  const int*   ocp   = (const int*)d_in[13];   // origin_context_pos
  const int*   xpp   = (const int*)d_in[15];   // x_pos
  float* out = (float*)d_out;

  char* ws = (char*)d_ws;
  // ---- overlay phase 1: ln1 | qkv | heads  (138,412,032 B) ----
  float* ln1 = (float*)(ws + 0);                              // 6144x1024 f32
  float* qkv = (float*)(ws + 25165824);                       // 6144x3072 f32
  bf16*  Qx  = (bf16*)(ws + 100663296);                       // 16x2048x64
  bf16*  Kx  = (bf16*)(ws + 104857600);
  bf16*  Vxt = (bf16*)(ws + 109051904);
  bf16*  Qc  = (bf16*)(ws + 113246208);                       // 16x4096x64
  bf16*  Kc  = (bf16*)(ws + 121634816);
  bf16*  Vct = (bf16*)(ws + 130023424);
  // ---- overlay phase 2 (reuses phase-1 space after attention): FFN ----
  float* h1  = (float*)(ws + 0);                              // 4096x2816 f32
  float* h3  = (float*)(ws + 46137344);
  float* hg  = (float*)(ws + 92274688);
  // ---- persistent ----
  float* attn = (float*)(ws + 138412032);                     // 4096x1024 f32
  float* proj = (float*)(ws + 155189248);
  float* xo   = (float*)(ws + 171966464);
  float* ln2  = (float*)(ws + 188743680);
  (void)in_sizes; (void)n_in; (void)out_size; (void)ws_size;

  // 1) LN1 over x (rows 0..2047) and context (rows 2048..6143)
  ln_kernel<<<dim3(NTOK), dim3(256), 0, stream>>>(x, ctx, LX, n1_g, n1_b, ln1);

  // 2) QKV GEMM: [6144,1024] x [1024,3072] + bias
  launch_gemm(ln1, qkv_w, qkv, qkv_b, nullptr, NTOK, 3 * D_MODEL, D_MODEL, stream);

  // 3) RoPE + head split + V-transpose, bf16
  rope_kernel<<<dim3(NTOK / 8, NHEADS), dim3(256), 0, stream>>>(
      qkv, Qx, Kx, Vxt, Qc, Kc, Vct, xpp, ocp);

  // 4) a1: Qx vs context K/V (non-causal)
  attn_kernel<<<dim3(LX / 64, NHEADS), dim3(128), 0, stream>>>(
      Qx, Kc, Vct, attn, LC, (long)LX * HD, (long)LC * HD, 0, 0);
  // 5) a2: Qx vs x K/V (causal), accumulated onto a1
  attn_kernel<<<dim3(LX / 64, NHEADS), dim3(128), 0, stream>>>(
      Qx, Kx, Vxt, attn, LX, (long)LX * HD, (long)LX * HD, 1, 1);
  // 6) a3: context-slice queries (rows 2048..4095 of Qc) vs context K/V
  attn_kernel<<<dim3(LX / 64, NHEADS), dim3(128), 0, stream>>>(
      Qc + (size_t)LX * HD, Kc, Vct, attn + (size_t)LX * D_MODEL,
      LC, (long)LC * HD, (long)LC * HD, 0, 0);

  // 7) output projection: [4096,1024] x [1024,1024] + bias
  launch_gemm(attn, out_w, proj, out_b, nullptr, NROWS2, D_MODEL, D_MODEL, stream);

  // 8) residual: xo = proj + {x | ctx_sel}
  add_res_kernel<<<dim3((NROWS2 * D_MODEL) / 256), dim3(256), 0, stream>>>(
      proj, x, ctx, xo);

  // 9) LN2
  ln_kernel<<<dim3(NROWS2), dim3(256), 0, stream>>>(xo, xo, NROWS2, n2_g, n2_b, ln2);

  // 10) FFN up projections (phase-1 workspace is free now)
  launch_gemm(ln2, w1, h1, nullptr, nullptr, NROWS2, HIDDEN, D_MODEL, stream);
  launch_gemm(ln2, w3, h3, nullptr, nullptr, NROWS2, HIDDEN, D_MODEL, stream);

  // 11) hg = silu(h1) * h3
  silu_mul_kernel<<<dim3((NROWS2 * HIDDEN) / 256), dim3(256), 0, stream>>>(h1, h3, hg);

  // 12) down projection + residual, straight into d_out (stacked [x_out;c_out])
  launch_gemm(hg, w2, out, nullptr, xo, NROWS2, D_MODEL, HIDDEN, stream);
}
